// AdditiveAttention_22608707847054
// MI455X (gfx1250) — matrix operand order repaired and checked by reading, not compiled, not run on hardware
//
#include <hip/hip_runtime.h>

typedef __attribute__((ext_vector_type(16))) _Float16 v16h;
typedef __attribute__((ext_vector_type(8)))  _Float16 v8h;
typedef __attribute__((ext_vector_type(8)))  float    v8f;
typedef __attribute__((ext_vector_type(4)))  float    v4f;

#define BB 2
#define LL 512
#define DD 512
#define HH 8
#define DEPTH 64

#if defined(__has_builtin)
# if __has_builtin(__builtin_amdgcn_tanhf)
#  define FAST_TANH(x) __builtin_amdgcn_tanhf(x)
# else
#  define FAST_TANH(x) tanhf(x)
# endif
#else
# define FAST_TANH(x) tanhf(x)
#endif

// ---------------- CDNA5 async global->LDS copy (ASYNCcnt path) ----------------
// global_load_async_to_lds_b128: VDST = VGPR holding LDS byte address,
// VADDR = 64-bit global address. Per ISA 10.2 the LDS offset is the low 32 bits
// of the flat shared address. 16B alignment required on both sides.
static __device__ inline void async_copy_b128(const float* gsrc, const float* ldst) {
  unsigned loff = (unsigned)(uintptr_t)ldst;
  unsigned long long gaddr = (unsigned long long)(uintptr_t)gsrc;
  asm volatile("global_load_async_to_lds_b128 %0, %1, off"
               :: "v"(loff), "v"(gaddr) : "memory");
}
static __device__ inline void wait_async_all() {
  asm volatile("s_wait_asynccnt 0x0" ::: "memory");
}

// ---------- WMMA fragment helpers (layouts per CDNA5 ISA 7.12.2) ----------

// A 16x32 f16: lane<16 -> row M=lane, K {0..7,16..23}; lane>=16 -> row M=lane-16, K {8..15,24..31}.
static __device__ inline v16h load_A_f32(const float* __restrict__ src, int ld) {
  const int lane = threadIdx.x & 31;
  const int row  = lane & 15;
  const int kb   = (lane >> 4) << 3;            // 0 or 8
  const float* r = src + row * ld + kb;         // 16B aligned by construction
  v4f x0 = *(const v4f*)(r);
  v4f x1 = *(const v4f*)(r + 4);
  v4f x2 = *(const v4f*)(r + 16);
  v4f x3 = *(const v4f*)(r + 20);
  v16h a;
#pragma unroll
  for (int j = 0; j < 4; ++j) {
    a[j]      = (_Float16)x0[j];
    a[j + 4]  = (_Float16)x1[j];
    a[j + 8]  = (_Float16)x2[j];
    a[j + 12] = (_Float16)x3[j];
  }
  return a;
}

static __device__ inline v16h load_A_f16(const _Float16* __restrict__ src, int ld) {
  const int lane = threadIdx.x & 31;
  const int row  = lane & 15;
  const int kb   = (lane >> 4) << 3;
  const _Float16* r = src + row * ld + kb;      // 16B aligned by construction
  v8h lo = *(const v8h*)(r);
  v8h hi = *(const v8h*)(r + 16);
  v16h a;
#pragma unroll
  for (int j = 0; j < 8; ++j) { a[j] = lo[j]; a[j + 8] = hi[j]; }
  return a;
}

// B 32x16 f16 from LDS tile t[32+][LDW]: lane<16 -> col N=lane, K {0..7,16..23}; lane>=16 -> K {8..15,24..31}.
template <int LDW>
static __device__ inline v16h load_B_lds(const float (*t)[LDW], int krow0, int n0) {
  const int lane = threadIdx.x & 31;
  const int col  = (lane & 15) + n0;
  const int kb   = (lane >> 4) << 3;            // 0 or 8
  v16h b;
#pragma unroll
  for (int j = 0; j < 16; ++j) b[j] = (_Float16)t[krow0 + kb + (j & 7) + ((j >> 3) << 4)][col];
  return b;
}

// B 32x16 f16 direct from global f32 (used only where there is no cross-wave reuse: Wo).
static __device__ inline v16h load_B_f32(const float* __restrict__ src, int ld) {
  const int lane = threadIdx.x & 31;
  const int col  = lane & 15;
  const int kb   = (lane >> 4) << 3;
  v16h b;
#pragma unroll
  for (int j = 0; j < 16; ++j) b[j] = (_Float16)src[(kb + (j & 7) + ((j >> 3) << 4)) * ld + col];
  return b;
}

static __device__ inline v8f wmma16(v16h a, v16h b, v8f c) {
  return __builtin_amdgcn_wmma_f32_16x16x32_f16(false, a, false, b, (short)0, c,
                                                false, false);
}

// ---------- Stage 1: qT = qh@Wq + bb ; kT = kh@Wk ----------
// Per-head [512,64]x[64,64]; W (16KB) staged once per block via async copies.
__global__ void transform_kernel(const float* __restrict__ query,
                                 const float* __restrict__ key,
                                 const float* __restrict__ Wq,
                                 const float* __restrict__ Wk,
                                 const float* __restrict__ bb,
                                 float* __restrict__ qT,
                                 float* __restrict__ kT) {
  __shared__ __align__(16) float Wl[64][68];   // stride 68: 16B-aligned rows
  const int tid  = threadIdx.x;                // 128 threads = 4 waves
  const int wave = tid >> 5;
  const int lane = tid & 31;
  const int bh   = blockIdx.y;                 // b*8+h
  const int b    = bh >> 3, h = bh & 7;
  const int isK  = blockIdx.z;
  const int row0 = blockIdx.x * 64 + wave * 16;

  const float* X = (isK ? key : query) + (size_t)(b * LL + row0) * DD + h * DEPTH;
  const float* W = (isK ? Wk : Wq) + h * DEPTH * DEPTH;
  float*       Y = (isK ? kT : qT) + (size_t)(bh * LL + row0) * DEPTH;

  // Stage W[64][64] into LDS: 1024 16B chunks over 128 threads.
#pragma unroll
  for (int c = tid; c < 1024; c += 128)
    async_copy_b128(W + c * 4, &Wl[c >> 4][(c & 15) * 4]);
  wait_async_all();
  __syncthreads();

  v8f acc[4] = {};
#pragma unroll
  for (int kk = 0; kk < DEPTH; kk += 32) {
    v16h a = load_A_f32(X + kk, DD);
#pragma unroll
    for (int nt = 0; nt < 4; ++nt)
      acc[nt] = wmma16(a, load_B_lds<68>(Wl, kk, nt * 16), acc[nt]);
  }
  const int n  = lane & 15;
  const int mb = (lane >> 4) << 3;
#pragma unroll
  for (int nt = 0; nt < 4; ++nt) {
    const float add = isK ? 0.f : bb[h * DEPTH + nt * 16 + n];
#pragma unroll
    for (int r = 0; r < 8; ++r)
      Y[(mb + r) * DEPTH + nt * 16 + n] = acc[nt][r] + add;
  }
}

// ---------- Stage 2: additive scores (dominant cost: 268M v_tanh_f32) ----------
__global__ void score_kernel(const float* __restrict__ qT,
                             const float* __restrict__ kT,
                             const float* __restrict__ va,
                             float* __restrict__ scores) {
  __shared__ __align__(16) float qs[16][68];
  __shared__ __align__(16) float ks[16][68];
  __shared__ float vs[DEPTH];
  const int bh = blockIdx.z;
  const int h  = bh & 7;
  const int q0 = blockIdx.x * 16;
  const int k0 = blockIdx.y * 16;
  const int tid = threadIdx.x;                 // 256 threads

  const float* qsrc = qT + (size_t)(bh * LL + q0) * DEPTH;
  const float* ksrc = kT + (size_t)(bh * LL + k0) * DEPTH;
  // 16x64 f32 tiles = 256 16B chunks each: one chunk per thread per tile.
  async_copy_b128(qsrc + tid * 4, &qs[tid >> 4][(tid & 15) * 4]);
  async_copy_b128(ksrc + tid * 4, &ks[tid >> 4][(tid & 15) * 4]);
  if (tid < DEPTH) vs[tid] = va[h * DEPTH + tid];
  wait_async_all();
  __syncthreads();

  const int q = tid >> 4, k = tid & 15;
  float s = 0.f;
#pragma unroll
  for (int e = 0; e < DEPTH; ++e)
    s = fmaf(FAST_TANH(qs[q][e] + ks[k][e]), vs[e], s);   // bb folded into qT
  scores[((size_t)bh * LL + (q0 + q)) * LL + (k0 + k)] = s;
}

// ---------- Stage 3: row softmax, emit f16 attn (one wave per 512-row) ----------
__global__ void softmax_kernel(const float* __restrict__ scores,
                               _Float16* __restrict__ attn) {
  const int wave = threadIdx.x >> 5;
  const int lane = threadIdx.x & 31;
  const int row  = blockIdx.x * 8 + wave;
  const float* src = scores + (size_t)row * LL;
  float v[16];
  float m = -3.4e38f;
#pragma unroll
  for (int i = 0; i < 16; ++i) { v[i] = src[lane + i * 32]; m = fmaxf(m, v[i]); }
#pragma unroll
  for (int off = 16; off > 0; off >>= 1) m = fmaxf(m, __shfl_xor(m, off, 32));
  float sum = 0.f;
#pragma unroll
  for (int i = 0; i < 16; ++i) { v[i] = __expf(v[i] - m); sum += v[i]; }
#pragma unroll
  for (int off = 16; off > 0; off >>= 1) sum += __shfl_xor(sum, off, 32);
  const float inv = 1.f / sum;
  _Float16* dst = attn + (size_t)row * LL;
#pragma unroll
  for (int i = 0; i < 16; ++i) dst[lane + i * 32] = (_Float16)(v[i] * inv);
}

// ---------- Stage 4: out = attn @ kh ----------
// All 8 waves in a block share the same (b,h): kh K-tile (32x64 f32, 8KB) is
// async-staged into LDS once per K step instead of 8x redundant global reads.
__global__ void av_kernel(const _Float16* __restrict__ attn,
                          const float* __restrict__ key,
                          float* __restrict__ merged) {
  __shared__ __align__(16) float khs[32][68];
  const int tid  = threadIdx.x;                // 256 threads
  const int wave = tid >> 5;
  const int lane = tid & 31;
  const int bh   = blockIdx.y;
  const int b    = bh >> 3, h = bh & 7;
  const int q0   = (blockIdx.x * 8 + wave) * 16;

  const _Float16* A = attn + ((size_t)bh * LL + q0) * LL;
  const float*    K = key + (size_t)b * LL * DD + h * DEPTH;

  v8f acc[4] = {};
  for (int kk = 0; kk < LL; kk += 32) {
    __syncthreads();                           // previous tile fully consumed
    // 32x64 f32 tile = 512 16B chunks over 256 threads (2 each).
#pragma unroll
    for (int c = tid; c < 512; c += 256)
      async_copy_b128(K + (size_t)(kk + (c >> 4)) * DD + (c & 15) * 4,
                      &khs[c >> 4][(c & 15) * 4]);
    wait_async_all();
    __syncthreads();

    v16h a = load_A_f16(A + kk, LL);
#pragma unroll
    for (int nt = 0; nt < 4; ++nt)
      acc[nt] = wmma16(a, load_B_lds<68>(khs, 0, nt * 16), acc[nt]);
  }
  const int n = lane & 15, mb = (lane >> 4) << 3;
  float* Y = merged + (size_t)(b * LL + q0) * DD + h * DEPTH;
#pragma unroll
  for (int nt = 0; nt < 4; ++nt)
#pragma unroll
    for (int r = 0; r < 8; ++r)
      Y[(mb + r) * DD + nt * 16 + n] = acc[nt][r];
}

// ---------- Stage 5: output projection merged[1024,512] @ Wo + bo ----------
// A panel (16 rows x 512, 32KB) shared by all 8 waves: async-staged once.
__global__ void out_kernel(const float* __restrict__ merged,
                           const float* __restrict__ Wo,
                           const float* __restrict__ bo,
                           float* __restrict__ out) {
  __shared__ __align__(16) float As[16][516];  // 516 floats = 129x16B rows
  const int tid  = threadIdx.x;                // 256 threads
  const int wave = tid >> 5;
  const int lane = tid & 31;
  const int m0   = blockIdx.x * 16;            // 1024 rows total
  const int n0   = wave * 64;                  // 8 waves cover 512 cols

  const float* Ag = merged + (size_t)m0 * DD;  // 16x512 contiguous
#pragma unroll
  for (int c = tid; c < 2048; c += 256)        // 2048 16B chunks
    async_copy_b128(Ag + c * 4, &As[c >> 7][(c & 127) * 4]);
  wait_async_all();
  __syncthreads();

  v8f acc[4] = {};
  for (int kk = 0; kk < DD; kk += 32) {
    // A fragment from LDS (vector ds loads, 16B aligned by construction).
    const int row = lane & 15;
    const int kb  = (lane >> 4) << 3;
    const float* r = &As[row][kk + kb];
    v4f x0 = *(const v4f*)(r);
    v4f x1 = *(const v4f*)(r + 4);
    v4f x2 = *(const v4f*)(r + 16);
    v4f x3 = *(const v4f*)(r + 20);
    v16h a;
#pragma unroll
    for (int j = 0; j < 4; ++j) {
      a[j] = (_Float16)x0[j]; a[j + 4] = (_Float16)x1[j];
      a[j + 8] = (_Float16)x2[j]; a[j + 12] = (_Float16)x3[j];
    }
#pragma unroll
    for (int nt = 0; nt < 4; ++nt)
      acc[nt] = wmma16(a, load_B_f32(Wo + (size_t)kk * DD + n0 + nt * 16, DD), acc[nt]);
  }
  const int n = lane & 15, mb = (lane >> 4) << 3;
#pragma unroll
  for (int nt = 0; nt < 4; ++nt) {
    const float bias = bo[n0 + nt * 16 + n];
#pragma unroll
    for (int r = 0; r < 8; ++r)
      out[(size_t)(m0 + mb + r) * DD + n0 + nt * 16 + n] = acc[nt][r] + bias;
  }
}

extern "C" void kernel_launch(void* const* d_in, const int* in_sizes, int n_in,
                              void* d_out, int out_size, void* d_ws, size_t ws_size,
                              hipStream_t stream) {
  const float* query = (const float*)d_in[0];
  const float* key   = (const float*)d_in[1];
  const float* Wq    = (const float*)d_in[2];
  const float* Wk    = (const float*)d_in[3];
  const float* va    = (const float*)d_in[4];
  const float* bb    = (const float*)d_in[5];
  const float* Wo    = (const float*)d_in[6];
  const float* bo    = (const float*)d_in[7];
  float* out = (float*)d_out;

  // Workspace layout (30 MB total, fully L2-resident on MI455X):
  char* ws = (char*)d_ws;
  float*    qT     = (float*)(ws);                  //  2 MB  [B,H,L,64]
  float*    kT     = (float*)(ws + (2u  << 20));    //  2 MB  [B,H,L,64]
  float*    scores = (float*)(ws + (4u  << 20));    // 16 MB  [B,H,L,L] f32
  _Float16* attn   = (_Float16*)(ws + (20u << 20)); //  8 MB  [B,H,L,L] f16
  float*    merged = (float*)(ws + (28u << 20));    //  2 MB  [B,L,D]

  transform_kernel<<<dim3(8, 16, 2),   128, 0, stream>>>(query, key, Wq, Wk, bb, qT, kT);
  score_kernel    <<<dim3(32, 32, 16), 256, 0, stream>>>(qT, kT, va, scores);
  softmax_kernel  <<<dim3(1024),       256, 0, stream>>>(scores, attn);
  av_kernel       <<<dim3(4, 16),      256, 0, stream>>>(attn, key, merged);
  out_kernel      <<<dim3(64),         256, 0, stream>>>(merged, Wo, bo, out);
}
